// EvolveGCN_1_layer_3719441678528
// MI455X (gfx1250) — matrix operand order repaired and checked by reading, not compiled, not run on hardware
//
#include <hip/hip_runtime.h>
#include <hip/hip_bf16.h>

// ---------------------------------------------------------------------------
// EvolveGCN (1 layer) for gfx1250 / MI455X.  wave32, WMMA f32_16x16x32_f16,
// fragment-swizzled operands (v16h contiguous per lane), async-to-LDS staging.
// T=8, N=50000, F0=H0=128, H1=64, NNZ=800000, E=1000000
// ---------------------------------------------------------------------------

#define T_ 8
#define N_ 50000
#define F0_ 128
#define H0_ 128
#define H1_ 64
#define NNZ_ 800000
#define E_ 1000000

typedef __attribute__((ext_vector_type(16))) _Float16 v16h;
typedef __attribute__((ext_vector_type(8)))  float    v8f;

// ---- WMMA ------------------------------------------------------------------

__device__ __forceinline__ v8f wmma_f16(v16h a, v16h b, v8f c) {
  return __builtin_amdgcn_wmma_f32_16x16x32_f16(
      /*neg_a=*/false, a, /*neg_b=*/false, b,
      /*c_mod=*/(short)0, c, /*reuse_a=*/false, /*reuse_b=*/false);
}

// Fast activations on the hardware transcendental path (v_exp/v_rcp).
__device__ __forceinline__ float fast_sigmoid(float x) {
  return __builtin_amdgcn_rcpf(1.f + __expf(-x));
}
__device__ __forceinline__ float fast_tanh(float x) {
  float e = __expf(2.f * x);
  return (e - 1.f) * __builtin_amdgcn_rcpf(e + 1.f);
}

// Fragment-swizzled layouts (one v16h per lane per fragment):
//   A frag (16x32 tile):  frag[(mt*KC + kc)*32 + lane]
//     inverse map of ISA 7.12.2 A layout:
//       element (m,k): mt=m>>4, l=m&15; kk=k&31; g=(kk>>3)&1;
//       h=(kk&7)+((kk>>4)<<3); lane=g*16+l
//   B frag (32x16 tile):  frag[(kc*NT + nt)*32 + lane]
//       element (k,n): kc=k>>5, g=(k>>3)&1, h=(k&7)+(((k>>4)&1)<<3); nt=n>>4, l=n&15;
//       lane=g*16+l

// ---- async-to-LDS (CDNA5, tracked by ASYNCcnt) -----------------------------

__device__ __forceinline__ void async_copy16(unsigned lds_addr, const void* g) {
  asm volatile("global_load_async_to_lds_b128 %0, %1, off"
               :: "v"(lds_addr), "v"((unsigned long long)(uintptr_t)g)
               : "memory");
}
__device__ __forceinline__ void async_wait0() {
  asm volatile("s_wait_asynccnt 0" ::: "memory");
}

// ---- swizzle kernels -------------------------------------------------------

// A operand: src f32 [128][128] row-major (M x K) -> A frags (MT=8, KC=4)
__global__ __launch_bounds__(256) void k_swizA(const float* src, _Float16* dst) {
  int i = blockIdx.x * 256 + threadIdx.x;
  if (i >= F0_ * F0_) return;
  int m = i >> 7, k = i & 127;
  int mt = m >> 4, l = m & 15;
  int kc = k >> 5, kk = k & 31;
  int g = (kk >> 3) & 1;
  int h = (kk & 7) + ((kk >> 4) << 3);
  int lane = g * 16 + l;
  dst[(size_t)((mt * 4 + kc) * 32 + lane) * 16 + h] = (_Float16)src[i];
}

// B operand: src f32 [K][Ncols] row-major -> B frags (KC=K/32, NT=Ncols/16)
__global__ __launch_bounds__(256) void k_swizB(const float* src, _Float16* dst,
                                               int K, int Ncols) {
  int i = blockIdx.x * 256 + threadIdx.x;
  if (i >= K * Ncols) return;
  int k = i / Ncols, n = i % Ncols;
  int NT = Ncols >> 4;
  int kc = k >> 5, g = (k >> 3) & 1, h = (k & 7) + (((k >> 4) & 1) << 3);
  int nt = n >> 4, l = n & 15;
  int lane = g * 16 + l;
  dst[(size_t)((kc * NT + nt) * 32 + lane) * 16 + h] = (_Float16)src[i];
}

// ---- small utility kernels -------------------------------------------------

__global__ __launch_bounds__(256) void k_init(const float* p, const float* Winit,
                                              float* phat, float* W0,
                                              int* counts, int* cursors) {
  __shared__ float red[256];
  int tid = threadIdx.x;
  float v = (tid < F0_) ? p[tid] * p[tid] : 0.f;
  red[tid] = v;
  __syncthreads();
  for (int s = 128; s > 0; s >>= 1) {
    if (tid < s) red[tid] += red[tid + s];
    __syncthreads();
  }
  float nrm = sqrtf(red[0]);
  if (tid < F0_) phat[tid] = p[tid] / nrm;
  for (int i = tid; i < F0_ * H0_; i += 256) W0[i] = Winit[i];
  if (tid < T_) { counts[tid] = 0; cursors[tid] = 0; }
}

__global__ __launch_bounds__(256) void k_zero(float* p, int n) {
  int i = blockIdx.x * 256 + threadIdx.x;
  if (i < n) p[i] = 0.f;
}

__global__ __launch_bounds__(256) void k_count(const int* et, int* counts) {
  int e = blockIdx.x * 256 + threadIdx.x;
  if (e < E_) atomicAdd(&counts[et[e]], 1);
}

__global__ void k_scan(const int* counts, int* offs) {
  if (threadIdx.x == 0) {
    int s = 0;
    for (int t = 0; t < T_; ++t) { offs[t] = s; s += counts[t]; }
    offs[T_] = s;
  }
}

__global__ __launch_bounds__(256) void k_fill(const int* et, int* cursors,
                                              const int* offs, int* elist) {
  int e = blockIdx.x * 256 + threadIdx.x;
  if (e < E_) {
    int t = et[e];
    int pos = atomicAdd(&cursors[t], 1);
    elist[offs[t] + pos] = e;
  }
}

__global__ __launch_bounds__(256) void k_copyW(const float* W, float* dst) {
  int i = blockIdx.x * 256 + threadIdx.x;
  if (i < F0_ * H0_) dst[i] = W[i];
}

// ---- per-step kernels ------------------------------------------------------

// y[n] = dot(X_t[n,:], p_hat)  — one wave per row, shuffle reduce.
__global__ __launch_bounds__(256) void k_score(const float* Xt, const float* ph,
                                               float* y) {
  int w = (blockIdx.x * 256 + threadIdx.x) >> 5;
  int lane = threadIdx.x & 31;
  if (w >= N_) return;
  const float4* x = (const float4*)(Xt + (size_t)w * F0_);
  const float4* p4 = (const float4*)ph;
  float4 a = x[lane], b = p4[lane];
  float s = a.x * b.x + a.y * b.y + a.z * b.z + a.w * b.w;
#pragma unroll
  for (int m = 16; m > 0; m >>= 1) s += __shfl_xor(s, m, 32);
  if (lane == 0) y[w] = s;
}

// top-128 by repeated argmax (single block, destroys y). Ties -> lowest index.
__global__ __launch_bounds__(1024) void k_topk(float* y, float* topv, int* topi) {
  __shared__ float sv[1024];
  __shared__ int   si[1024];
  int tid = threadIdx.x;
  for (int j = 0; j < H0_; ++j) {
    float bv = -__builtin_inff();
    int bi = N_;
    for (int n = tid; n < N_; n += 1024) {
      float v = y[n];
      if (v > bv || (v == bv && n < bi)) { bv = v; bi = n; }
    }
    sv[tid] = bv; si[tid] = bi;
    __syncthreads();
    for (int s = 512; s > 0; s >>= 1) {
      if (tid < s) {
        if (sv[tid + s] > sv[tid] ||
            (sv[tid + s] == sv[tid] && si[tid + s] < si[tid])) {
          sv[tid] = sv[tid + s]; si[tid] = si[tid + s];
        }
      }
      __syncthreads();
    }
    if (tid == 0) {
      topv[j] = sv[0]; topi[j] = si[0];
      y[si[0]] = -__builtin_inff();
    }
    __syncthreads();
  }
}

// Z[f,j] = X_t[idx[j], f] * topv[j] written directly in B-frag layout (NT=8)
__global__ __launch_bounds__(128) void k_buildZ(const float* Xt, const int* topi,
                                                const float* topv, _Float16* Zb) {
  int j = blockIdx.x;   // column (n)
  int f = threadIdx.x;  // row (k)
  float v = Xt[(size_t)topi[j] * F0_ + f] * topv[j];
  int kc = f >> 5, g = (f >> 3) & 1, h = (f & 7) + (((f >> 4) & 1) << 3);
  int nt = j >> 4, l = j & 15;
  int lane = g * 16 + l;
  Zb[(size_t)((kc * 8 + nt) * 32 + lane) * 16 + h] = (_Float16)v;
}

// Gate = sigmoid(Wg @ Z + Ug @ W + Bg).  blockIdx 0 -> Z gate, 1 -> R gate.
// All operands pre-swizzled: A frags (8x4), B frags (4x8). 8 waves.
__global__ __launch_bounds__(256) void k_gates(
    const v16h* WgA0, const v16h* UgA0, const float* Bg0, float* Out0,
    const v16h* WgA1, const v16h* UgA1, const float* Bg1, float* Out1,
    const v16h* Zb, const v16h* Wb) {
  const v16h *WgA, *UgA; const float* Bg; float* Out;
  if (blockIdx.x == 0) { WgA = WgA0; UgA = UgA0; Bg = Bg0; Out = Out0; }
  else                 { WgA = WgA1; UgA = UgA1; Bg = Bg1; Out = Out1; }
  int wave = threadIdx.x >> 5, lane = threadIdx.x & 31;
  int mt = wave, m0 = wave * 16;
  v16h aW[4], aU[4];
#pragma unroll
  for (int kc = 0; kc < 4; ++kc) {
    aW[kc] = WgA[(mt * 4 + kc) * 32 + lane];
    aU[kc] = UgA[(mt * 4 + kc) * 32 + lane];
  }
  for (int nt = 0; nt < 8; ++nt) {
    int n0 = nt * 16;
    v8f acc = {};
#pragma unroll
    for (int kc = 0; kc < 4; ++kc)
      acc = wmma_f16(aW[kc], Zb[(kc * 8 + nt) * 32 + lane], acc);
#pragma unroll
    for (int kc = 0; kc < 4; ++kc)
      acc = wmma_f16(aU[kc], Wb[(kc * 8 + nt) * 32 + lane], acc);
    int l = lane & 15, g = lane >> 4;
#pragma unroll
    for (int r = 0; r < 8; ++r) {
      int m = m0 + g * 8 + r, n = n0 + l;
      Out[m * H0_ + n] = fast_sigmoid(acc[r] + Bg[m * H0_ + n]);
    }
  }
}

// RW = R * W (elementwise)
__global__ __launch_bounds__(256) void k_rw(const float* R, const float* Wc, float* RW) {
  int i = blockIdx.x * 256 + threadIdx.x;
  if (i < F0_ * H0_) RW[i] = R[i] * Wc[i];
}

// Ht = tanh(Wh@Z + Uh@RW + Bh);  Wn = (1-Zg)*Wc + Zg*Ht.   One block, 8 waves.
__global__ __launch_bounds__(256) void k_htupd(
    const v16h* WhA, const v16h* UhA, const float* Bh,
    const v16h* Zb, const v16h* RWb, const float* Zg,
    const float* Wc, float* Wn) {
  int wave = threadIdx.x >> 5, lane = threadIdx.x & 31;
  int mt = wave, m0 = wave * 16;
  v16h aW[4], aU[4];
#pragma unroll
  for (int kc = 0; kc < 4; ++kc) {
    aW[kc] = WhA[(mt * 4 + kc) * 32 + lane];
    aU[kc] = UhA[(mt * 4 + kc) * 32 + lane];
  }
  for (int nt = 0; nt < 8; ++nt) {
    int n0 = nt * 16;
    v8f acc = {};
#pragma unroll
    for (int kc = 0; kc < 4; ++kc)
      acc = wmma_f16(aW[kc], Zb[(kc * 8 + nt) * 32 + lane], acc);
#pragma unroll
    for (int kc = 0; kc < 4; ++kc)
      acc = wmma_f16(aU[kc], RWb[(kc * 8 + nt) * 32 + lane], acc);
    int l = lane & 15, g = lane >> 4;
#pragma unroll
    for (int r = 0; r < 8; ++r) {
      int m = m0 + g * 8 + r, n = n0 + l;
      int idx = m * H0_ + n;
      float ht = fast_tanh(acc[r] + Bh[idx]);
      float z = Zg[idx];
      Wn[idx] = (1.f - z) * Wc[idx] + z * ht;
    }
  }
}

// SpMM scatter: AH[rows[i],:] += vals[i] * X_t[cols[i],:]   (one wave per nnz)
__global__ __launch_bounds__(256) void k_scatter(const int* rows, const int* cols,
                                                 const float* vals, const float* Xt,
                                                 float* AH) {
  int w = (blockIdx.x * 256 + threadIdx.x) >> 5;
  int lane = threadIdx.x & 31;
  if (w >= NNZ_) return;
  int r = rows[w], c = cols[w];
  float v = vals[w];
  const float4* xs = (const float4*)(Xt + (size_t)c * F0_);
  float4 x = xs[lane];
  float* dst = AH + (size_t)r * F0_ + lane * 4;
  atomicAdd(dst + 0, v * x.x);
  atomicAdd(dst + 1, v * x.y);
  atomicAdd(dst + 2, v * x.z);
  atomicAdd(dst + 3, v * x.w);
}

// Y_t = AH @ Wn : 64 rows/block; W frags async-copied to LDS, A swizzled in LDS.
#define YROWS 64
__global__ __launch_bounds__(256) void k_ygemm(const float* AH, const v16h* Wb,
                                               float* Yt) {
  __shared__ v16h sW[4 * 8 * 32];   // 32 KB : B frags (KC=4, NT=8)
  __shared__ v16h sA[4 * 4 * 32];   // 16 KB : A frags (MT=4, KC=4)
  int tid = threadIdx.x;
  // async bulk copy of pre-swizzled W fragments into LDS
  {
    const char* src = (const char*)Wb;
    char* dst = (char*)sW;
    for (int i = tid; i < 2048; i += 256)
      async_copy16((unsigned)(uintptr_t)(dst + i * 16), src + i * 16);
  }
  // gather + swizzle A rows into LDS fragment layout
  int rbase = blockIdx.x * YROWS;
  _Float16* sAh = (_Float16*)sA;
  for (int i = tid; i < YROWS * F0_; i += 256) {
    int lm = i >> 7, k = i & 127;
    int r = rbase + lm;
    float v = (r < N_) ? AH[(size_t)r * F0_ + k] : 0.f;
    int mt = lm >> 4, l = lm & 15;
    int kc = k >> 5, kk = k & 31;
    int g = (kk >> 3) & 1;
    int h = (kk & 7) + ((kk >> 4) << 3);
    int lane = g * 16 + l;
    sAh[(size_t)((mt * 4 + kc) * 32 + lane) * 16 + h] = (_Float16)v;
  }
  async_wait0();
  __syncthreads();
  int wave = tid >> 5, lane = tid & 31;
  int mt = wave >> 1;
  int m0 = mt * 16;
  int nthalf = (wave & 1) * 4;
  v16h aF[4];
#pragma unroll
  for (int kc = 0; kc < 4; ++kc) aF[kc] = sA[(mt * 4 + kc) * 32 + lane];
  for (int ntl = 0; ntl < 4; ++ntl) {
    int nt = nthalf + ntl;
    int n0 = nt * 16;
    v8f acc = {};
#pragma unroll
    for (int kc = 0; kc < 4; ++kc)
      acc = wmma_f16(aF[kc], sW[(kc * 8 + nt) * 32 + lane], acc);
    int l = lane & 15, g = lane >> 4;
#pragma unroll
    for (int r = 0; r < 8; ++r) {
      int row = rbase + m0 + g * 8 + r;
      if (row < N_) Yt[(size_t)row * H0_ + n0 + l] = acc[r];
    }
  }
}

// Edge output: out[e,:] = Yt[src]@U[0:128] + Yt[trg]@U[128:256]
// 32 edges/block; U frags async-copied to LDS; 8 waves = (2 edge-grp x 4 col).
__global__ __launch_bounds__(256) void k_edge(int t, const int* elist, const int* offs,
                                              const int* counts, const int* esrc,
                                              const int* etrg, const float* Yt,
                                              const v16h* Ub, float* out) {
  int cnt = counts[t];
  int base = blockIdx.x * 32;
  if (base >= cnt) return;
  __shared__ v16h sU[8 * 4 * 32];   // 32 KB : B frags (KC=8, NT=4)
  __shared__ v16h sA[2 * 8 * 32];   // 16 KB : A frags (MT=2, KC=8)
  __shared__ int sEid[32];
  int tid = threadIdx.x;
  int off = offs[t];
  if (tid < 32) {
    int ei = base + tid;
    sEid[tid] = (ei < cnt) ? elist[off + ei] : -1;
  }
  {
    const char* src = (const char*)Ub;
    char* dst = (char*)sU;
    for (int i = tid; i < 2048; i += 256)
      async_copy16((unsigned)(uintptr_t)(dst + i * 16), src + i * 16);
  }
  __syncthreads();   // sEid visible
  _Float16* sAh = (_Float16*)sA;
  for (int i = tid; i < 32 * 256; i += 256) {
    int slot = i >> 8, k = i & 255;
    int e = sEid[slot];
    float v = 0.f;
    if (e >= 0) {
      int node = (k < H0_) ? esrc[e] : etrg[e];
      v = Yt[(size_t)node * H0_ + (k & 127)];
    }
    int mt = slot >> 4, l = slot & 15;
    int kc = k >> 5, kk = k & 31;
    int g = (kk >> 3) & 1;
    int h = (kk & 7) + ((kk >> 4) << 3);
    int lane = g * 16 + l;
    sAh[(size_t)((mt * 8 + kc) * 32 + lane) * 16 + h] = (_Float16)v;
  }
  async_wait0();
  __syncthreads();
  int wave = tid >> 5, lane = tid & 31;
  int mt = wave >> 2;              // edge group
  int nt = wave & 3;               // output col tile
  int m0 = mt * 16, n0 = nt * 16;
  v8f acc = {};
#pragma unroll
  for (int kc = 0; kc < 8; ++kc)
    acc = wmma_f16(sA[(mt * 8 + kc) * 32 + lane],
                   sU[(kc * 4 + nt) * 32 + lane], acc);
  int l = lane & 15, g = lane >> 4;
#pragma unroll
  for (int r = 0; r < 8; ++r) {
    int slot = m0 + g * 8 + r;
    int e = sEid[slot];
    if (e >= 0) out[(size_t)e * H1_ + n0 + l] = acc[r];
  }
}

// ---------------------------------------------------------------------------

static inline char* wsalloc(char*& cur, size_t bytes) {
  char* p = cur;
  cur += (bytes + 255) & ~(size_t)255;
  return p;
}

extern "C" void kernel_launch(void* const* d_in, const int* in_sizes, int n_in,
                              void* d_out, int out_size, void* d_ws, size_t ws_size,
                              hipStream_t stream) {
  const float* X       = (const float*)d_in[0];
  const int*   A_rows  = (const int*)d_in[1];
  const int*   A_cols  = (const int*)d_in[2];
  const float* A_val   = (const float*)d_in[3];
  const int*   e_time  = (const int*)d_in[4];
  const int*   e_src   = (const int*)d_in[5];
  const int*   e_trg   = (const int*)d_in[6];
  const float* p       = (const float*)d_in[7];
  const float* W_Z     = (const float*)d_in[8];
  const float* U_Z     = (const float*)d_in[9];
  const float* B_Z     = (const float*)d_in[10];
  const float* W_R     = (const float*)d_in[11];
  const float* U_R     = (const float*)d_in[12];
  const float* B_R     = (const float*)d_in[13];
  const float* W_H     = (const float*)d_in[14];
  const float* U_H     = (const float*)d_in[15];
  const float* B_H     = (const float*)d_in[16];
  const float* W_init  = (const float*)d_in[17];
  const float* U       = (const float*)d_in[18];
  float* out = (float*)d_out;                       // [E,H1] then W_fin [F0,H0]

  const size_t MAT = (size_t)F0_ * H0_;             // 16384
  char* cur = (char*)d_ws;
  float*     phat   = (float*)wsalloc(cur, F0_ * 4);
  float*     y      = (float*)wsalloc(cur, (size_t)N_ * 4);
  float*     topv   = (float*)wsalloc(cur, H0_ * 4);
  int*       topi   = (int*)wsalloc(cur, H0_ * 4);
  float*     Zg     = (float*)wsalloc(cur, MAT * 4);
  float*     Rg     = (float*)wsalloc(cur, MAT * 4);
  float*     RW     = (float*)wsalloc(cur, MAT * 4);
  float*     Wbuf0  = (float*)wsalloc(cur, MAT * 4);
  float*     Wbuf1  = (float*)wsalloc(cur, MAT * 4);
  // fragment-swizzled operands (f16)
  _Float16*  WzA    = (_Float16*)wsalloc(cur, MAT * 2);
  _Float16*  UzA    = (_Float16*)wsalloc(cur, MAT * 2);
  _Float16*  WrA    = (_Float16*)wsalloc(cur, MAT * 2);
  _Float16*  UrA    = (_Float16*)wsalloc(cur, MAT * 2);
  _Float16*  WhA    = (_Float16*)wsalloc(cur, MAT * 2);
  _Float16*  UhA    = (_Float16*)wsalloc(cur, MAT * 2);
  _Float16*  Zb     = (_Float16*)wsalloc(cur, MAT * 2);
  _Float16*  Wb     = (_Float16*)wsalloc(cur, MAT * 2);
  _Float16*  RWb    = (_Float16*)wsalloc(cur, MAT * 2);
  _Float16*  Ub     = (_Float16*)wsalloc(cur, (size_t)2 * H0_ * H1_ * 2);
  int*       counts = (int*)wsalloc(cur, T_ * 4);
  int*       offs   = (int*)wsalloc(cur, (T_ + 1) * 4);
  int*       cursors= (int*)wsalloc(cur, T_ * 4);
  int*       elist  = (int*)wsalloc(cur, (size_t)E_ * 4);
  float*     AH     = (float*)wsalloc(cur, (size_t)N_ * F0_ * 4);
  float*     Yt     = (float*)wsalloc(cur, (size_t)N_ * H0_ * 4);
  float*     Wbufs[2] = {Wbuf0, Wbuf1};

  // one-time setup
  k_init<<<1, 256, 0, stream>>>(p, W_init, phat, Wbuf0, counts, cursors);
  k_count<<<(E_ + 255) / 256, 256, 0, stream>>>(e_time, counts);
  k_scan<<<1, 32, 0, stream>>>(counts, offs);
  k_fill<<<(E_ + 255) / 256, 256, 0, stream>>>(e_time, cursors, offs, elist);
  k_swizA<<<64, 256, 0, stream>>>(W_Z, WzA);
  k_swizA<<<64, 256, 0, stream>>>(U_Z, UzA);
  k_swizA<<<64, 256, 0, stream>>>(W_R, WrA);
  k_swizA<<<64, 256, 0, stream>>>(U_R, UrA);
  k_swizA<<<64, 256, 0, stream>>>(W_H, WhA);
  k_swizA<<<64, 256, 0, stream>>>(U_H, UhA);
  k_swizB<<<64, 256, 0, stream>>>(U, Ub, 2 * H0_, H1_);
  k_swizB<<<64, 256, 0, stream>>>(W_init, Wb, F0_, H0_);   // W(0) as B operand

  int curW = 0;
  for (int t = 0; t < T_; ++t) {
    const float* Xt    = X + (size_t)t * N_ * F0_;
    const int*   rows  = A_rows + (size_t)t * NNZ_;
    const int*   cols  = A_cols + (size_t)t * NNZ_;
    const float* vals  = A_val + (size_t)t * NNZ_;
    const float* Wc    = Wbufs[curW];
    float*       Wn    = Wbufs[curW ^ 1];

    k_score<<<(N_ * 32 + 255) / 256, 256, 0, stream>>>(Xt, phat, y);
    k_topk<<<1, 1024, 0, stream>>>(y, topv, topi);
    k_buildZ<<<H0_, F0_, 0, stream>>>(Xt, topi, topv, Zb);
    k_gates<<<2, 256, 0, stream>>>((const v16h*)WzA, (const v16h*)UzA, B_Z, Zg,
                                   (const v16h*)WrA, (const v16h*)UrA, B_R, Rg,
                                   (const v16h*)Zb, (const v16h*)Wb);
    k_rw<<<(int)((MAT + 255) / 256), 256, 0, stream>>>(Rg, Wc, RW);
    k_swizB<<<64, 256, 0, stream>>>(RW, RWb, F0_, H0_);
    k_htupd<<<1, 256, 0, stream>>>((const v16h*)WhA, (const v16h*)UhA, B_H,
                                   (const v16h*)Zb, (const v16h*)RWb, Zg, Wc, Wn);
    k_swizB<<<64, 256, 0, stream>>>(Wn, Wb, F0_, H0_);     // = W_new, reused next t

    k_zero<<<((int)((size_t)N_ * F0_ + 255)) / 256, 256, 0, stream>>>(AH, N_ * F0_);
    k_scatter<<<(NNZ_ * 32 + 255) / 256, 256, 0, stream>>>(rows, cols, vals, Xt, AH);
    k_ygemm<<<(N_ + YROWS - 1) / YROWS, 256, 0, stream>>>(AH, (const v16h*)Wb, Yt);

    k_edge<<<(E_ + 31) / 32, 256, 0, stream>>>(t, elist, offs, counts,
                                               e_src, e_trg, Yt, (const v16h*)Ub, out);
    curW ^= 1;
  }

  // W_fin tail of d_out
  k_copyW<<<(F0_ * H0_ + 255) / 256, 256, 0, stream>>>(Wbufs[curW],
                                                       out + (size_t)E_ * H1_);
}
